// LinearAttention_20486994002481
// MI455X (gfx1250) — matrix operand order repaired and checked by reading, not compiled, not run on hardware
//
#include <hip/hip_runtime.h>
#include <hip/hip_bf16.h>

// ---------------------------------------------------------------------------
// Linear attention for MI455X (gfx1250, wave32, WMMA).
//   All 4 big GEMMs + both per-head einsums run on v_wmma_f32_16x16x32_f16.
//   f16 operand fragments (A/B: this file's contiguous k order; C/D per CDNA5 ISA 7.12.2):
//     A (16x32): lane L -> row M=L&15; halves = K[(L>>4)*16 .. +16) contiguous.
//     B (32x16): lane L -> col N=L&15; halves = K[(L>>4)*16 .. +16) contiguous, same order as A (not the ISA table).
//   C/D (16x16 f32): VGPR r, lane L -> M = r + 8*(L>>4), N = L&15.
// Inner GEMM loops use a ping-pong (unroll-2) software pipeline: two named
// fragment buffers X/Y, each refilled right after being consumed, so loads
// stay a full step ahead of their WMMAs *and* the register allocator keeps
// each buffer in fixed VGPRs (no loop-carried v_mov copies).
// Workspace peak ~130 MB with reuse.
// ---------------------------------------------------------------------------

typedef __attribute__((ext_vector_type(16))) _Float16 v16h;
typedef __attribute__((ext_vector_type(8)))  float    v8f;
typedef __attribute__((ext_vector_type(4)))  float    f32x4;

static constexpr int NB   = 4;       // batch
static constexpr int TT   = 4096;    // tokens per batch
static constexpr int NROW = NB * TT; // 16384
static constexpr int DIM  = 512;
static constexpr int H    = 8;
static constexpr int D    = 64;

#define DEV __device__ __forceinline__

DEV v16h ldfragA(const _Float16* p) {
  // p already offset by (lane>>4)*16; halves [0..15] contiguous
  union { v16h v; f32x4 q[2]; } u;
  u.q[0] = *reinterpret_cast<const f32x4*>(p);
  u.q[1] = *reinterpret_cast<const f32x4*>(p + 8);
  return u.v;
}

DEV v16h ldfragB(const _Float16* p) {
  // p already offset by (lane>>4)*16; halves [0..15] contiguous
  union { v16h v; f32x4 q[2]; } u;
  u.q[0] = reinterpret_cast<const f32x4*>(p)[0];
  u.q[1] = reinterpret_cast<const f32x4*>(p)[1];
  return u.v;
}

DEV v8f wmma16(v16h a, v16h b, v8f c) {
  return __builtin_amdgcn_wmma_f32_16x16x32_f16(false, a, false, b, (short)0, c,
                                                false, false);
}

// ---------------------------------------------------------------- cast f32->f16
__global__ void k_cast_f16(const float* __restrict__ in, _Float16* __restrict__ out,
                           int n4) {
  int i = blockIdx.x * blockDim.x + threadIdx.x;
  if (i >= n4) return;
  f32x4 v = reinterpret_cast<const f32x4*>(in)[i];
  union { _Float16 h[4]; unsigned long long u64; } u;
  u.h[0] = (_Float16)v.x; u.h[1] = (_Float16)v.y;
  u.h[2] = (_Float16)v.z; u.h[3] = (_Float16)v.w;
  reinterpret_cast<unsigned long long*>(out)[i] = u.u64;
}

// ------------------------------------------------- GEMM: out = A @ W^T + bias
// A: [NROW, DIM] f16 row-major.  W: [DIM, DIM] cast to f16 row-major; since
// out[n,m] = sum_k A[n,k]*W[m,k], B-fragment lane n reads W row n contiguously
// -> no transpose needed anywhere.
// Block = 256 thr (8 waves): 128 rows x 64 cols; wave = 16 rows x 64 cols.
__global__ void k_gemm_f16(const _Float16* __restrict__ A,
                           const _Float16* __restrict__ W,
                           const float* __restrict__ bias,
                           float* __restrict__ outF,
                           _Float16* __restrict__ outH) {
  const int lane = threadIdx.x & 31;
  const int wave = threadIdx.x >> 5;
  const int row0 = blockIdx.x * 128 + wave * 16;
  const int c0   = blockIdx.y * 64;
  const int r    = lane & 15;
  const int kA   = (lane >> 4) * 16;
  const int kB   = (lane >> 4) * 16;
  const _Float16* pa = A + (size_t)(row0 + r) * DIM + kA;
  const _Float16* pb = W + (size_t)(c0 + r) * DIM + kB;

  v8f acc[4] = {};
  // ping-pong pipeline: X holds step k-64, Y holds step k-32
  v16h aX  = ldfragA(pa);
  v16h bX0 = ldfragB(pb + 0 * 16 * DIM);
  v16h bX1 = ldfragB(pb + 1 * 16 * DIM);
  v16h bX2 = ldfragB(pb + 2 * 16 * DIM);
  v16h bX3 = ldfragB(pb + 3 * 16 * DIM);
  v16h aY  = ldfragA(pa + 32);
  v16h bY0 = ldfragB(pb + 0 * 16 * DIM + 32);
  v16h bY1 = ldfragB(pb + 1 * 16 * DIM + 32);
  v16h bY2 = ldfragB(pb + 2 * 16 * DIM + 32);
  v16h bY3 = ldfragB(pb + 3 * 16 * DIM + 32);
#pragma unroll 1
  for (int k0 = 64; k0 < DIM; k0 += 64) {
    __builtin_prefetch(pa + k0 + 512, 0, 1);   // global_prefetch_b8, A stream
    acc[0] = wmma16(aX, bX0, acc[0]);
    acc[1] = wmma16(aX, bX1, acc[1]);
    acc[2] = wmma16(aX, bX2, acc[2]);
    acc[3] = wmma16(aX, bX3, acc[3]);
    aX  = ldfragA(pa + k0);
    bX0 = ldfragB(pb + 0 * 16 * DIM + k0);
    bX1 = ldfragB(pb + 1 * 16 * DIM + k0);
    bX2 = ldfragB(pb + 2 * 16 * DIM + k0);
    bX3 = ldfragB(pb + 3 * 16 * DIM + k0);
    acc[0] = wmma16(aY, bY0, acc[0]);
    acc[1] = wmma16(aY, bY1, acc[1]);
    acc[2] = wmma16(aY, bY2, acc[2]);
    acc[3] = wmma16(aY, bY3, acc[3]);
    aY  = ldfragA(pa + k0 + 32);
    bY0 = ldfragB(pb + 0 * 16 * DIM + k0 + 32);
    bY1 = ldfragB(pb + 1 * 16 * DIM + k0 + 32);
    bY2 = ldfragB(pb + 2 * 16 * DIM + k0 + 32);
    bY3 = ldfragB(pb + 3 * 16 * DIM + k0 + 32);
  }
  // tail: X = step DIM-64, Y = step DIM-32
  acc[0] = wmma16(aX, bX0, acc[0]);
  acc[1] = wmma16(aX, bX1, acc[1]);
  acc[2] = wmma16(aX, bX2, acc[2]);
  acc[3] = wmma16(aX, bX3, acc[3]);
  acc[0] = wmma16(aY, bY0, acc[0]);
  acc[1] = wmma16(aY, bY1, acc[1]);
  acc[2] = wmma16(aY, bY2, acc[2]);
  acc[3] = wmma16(aY, bY3, acc[3]);

  const int rTop = (lane >> 4) * 8;
#pragma unroll
  for (int rr = 0; rr < 8; ++rr) {
    const int row = row0 + rr + rTop;
#pragma unroll
    for (int j = 0; j < 4; ++j) {
      const int col = c0 + j * 16 + r;
      const float v = acc[j][rr] + bias[col];
      if (outH) outH[(size_t)row * DIM + col] = (_Float16)v;
      else      outF[(size_t)row * DIM + col] = v;
    }
  }
}

// -------------------------------------- softmax over D=64 per (token, head)
// One wave per row-head; lane holds 2 consecutive columns; shuffle reductions.
__global__ void k_softmax_head(const float* __restrict__ P,
                               _Float16* __restrict__ outh) {
  const int lane = threadIdx.x & 31;
  const int wave = threadIdx.x >> 5;
  const int gid  = blockIdx.x * 8 + wave;   // row*H + h
  const int row  = gid >> 3;
  const int h    = gid & 7;
  const float* p = P + (size_t)row * DIM + h * D + lane * 2;
  const float v0 = p[0], v1 = p[1];
  float m = fmaxf(v0, v1);
#pragma unroll
  for (int off = 16; off > 0; off >>= 1) m = fmaxf(m, __shfl_xor(m, off, 32));
  const float e0 = __expf(v0 - m), e1 = __expf(v1 - m);
  float s = e0 + e1;
#pragma unroll
  for (int off = 16; off > 0; off >>= 1) s += __shfl_xor(s, off, 32);
  const float inv = 1.0f / s;
  _Float16* o = outh + (size_t)row * DIM + h * D + lane * 2;
  o[0] = (_Float16)(e0 * inv);
  o[1] = (_Float16)(e1 * inv);
}

// ------------------- per-head 64x64 tile transpose: [t, d] -> [bh][d][t] f16
// Coalesced both ways through LDS (row stride padded to 80 halves = 160B,
// keeps 16B-aligned b128 LDS accesses).
__global__ void k_transpose_head(const _Float16* __restrict__ src,
                                 _Float16* __restrict__ dst) {
  __shared__ _Float16 tile[64][80];
  const int t0    = blockIdx.x * 64;      // NROW/64 blocks; never straddles batch
  const int h     = blockIdx.y;
  const int b     = t0 / TT;
  const int tloc0 = t0 - b * TT;
  const int bh    = b * H + h;
  const int r = threadIdx.x >> 3;         // 0..31
  const int c = (threadIdx.x & 7) * 8;    // 0,8,..,56
#pragma unroll
  for (int rr = r; rr < 64; rr += 32) {
    *reinterpret_cast<f32x4*>(&tile[rr][c]) =
        *reinterpret_cast<const f32x4*>(&src[(size_t)(t0 + rr) * DIM + h * D + c]);
  }
  __syncthreads();
#pragma unroll
  for (int dd = r; dd < 64; dd += 32) {
    union { _Float16 hh[8]; f32x4 q; } u;
#pragma unroll
    for (int i = 0; i < 8; ++i) u.hh[i] = tile[c + i][dd];
    *reinterpret_cast<f32x4*>(&dst[((size_t)bh * D + dd) * TT + tloc0 + c]) = u.q;
  }
}

// -------------------------------- k_sum[bh*64+d] = sum_t khT[bh][d][t]
__global__ void k_ksum(const _Float16* __restrict__ khT, float* __restrict__ ksum) {
  const int lane = threadIdx.x & 31;
  const int wave = threadIdx.x >> 5;
  const int gid  = blockIdx.x * 8 + wave;   // 0..NB*H*D-1
  const _Float16* p = khT + (size_t)gid * TT;
  float s = 0.f;
  for (int i = lane; i < TT; i += 32) s += (float)p[i];
#pragma unroll
  for (int off = 16; off > 0; off >>= 1) s += __shfl_xor(s, off, 32);
  if (lane == 0) ksum[gid] = s;
}

// -------------------------------- Dinv[row*H+h] = 1 / <q_rowhead, ksum_bh>
__global__ void k_dinv(const _Float16* __restrict__ qh, const float* __restrict__ ksum,
                       float* __restrict__ Dinv) {
  const int lane = threadIdx.x & 31;
  const int wave = threadIdx.x >> 5;
  const int gid  = blockIdx.x * 8 + wave;
  const int row  = gid >> 3;
  const int h    = gid & 7;
  const int b    = row / TT;
  const float*    ks = ksum + ((size_t)b * H + h) * D + lane * 2;
  const _Float16* q  = qh + (size_t)row * DIM + h * D + lane * 2;
  float s = (float)q[0] * ks[0] + (float)q[1] * ks[1];
#pragma unroll
  for (int off = 16; off > 0; off >>= 1) s += __shfl_xor(s, off, 32);
  if (lane == 0) Dinv[gid] = 1.0f / s;
}

// --------------- context^T[bh][e][d] = sum_t k[t,d] v[t,e]  (K = 4096, WMMA)
// A = khT rows (d-major), B lane e = vhT row e (t contiguous) -> coalesced.
__global__ void k_context(const _Float16* __restrict__ khT,
                          const _Float16* __restrict__ vhT,
                          _Float16* __restrict__ ctxT) {
  const int lane = threadIdx.x & 31;
  const int wave = threadIdx.x >> 5;  // 4 waves: d-strips
  const int bh   = blockIdx.x;        // NB*H
  const int d0   = wave * 16;
  const int r    = lane & 15;
  const int kA   = (lane >> 4) * 16;
  const int kB   = (lane >> 4) * 16;
  const _Float16* pa = khT + ((size_t)bh * D + d0 + r) * TT + kA;
  const _Float16* pb = vhT + ((size_t)bh * D + r) * TT + kB;
  v8f acc[4] = {};
  v16h aX  = ldfragA(pa);
  v16h bX0 = ldfragB(pb + (size_t)(0 * 16) * TT);
  v16h bX1 = ldfragB(pb + (size_t)(1 * 16) * TT);
  v16h bX2 = ldfragB(pb + (size_t)(2 * 16) * TT);
  v16h bX3 = ldfragB(pb + (size_t)(3 * 16) * TT);
  v16h aY  = ldfragA(pa + 32);
  v16h bY0 = ldfragB(pb + (size_t)(0 * 16) * TT + 32);
  v16h bY1 = ldfragB(pb + (size_t)(1 * 16) * TT + 32);
  v16h bY2 = ldfragB(pb + (size_t)(2 * 16) * TT + 32);
  v16h bY3 = ldfragB(pb + (size_t)(3 * 16) * TT + 32);
#pragma unroll 1
  for (int t0 = 64; t0 < TT; t0 += 64) {
    __builtin_prefetch(pa + t0 + 512, 0, 1);
    acc[0] = wmma16(aX, bX0, acc[0]);
    acc[1] = wmma16(aX, bX1, acc[1]);
    acc[2] = wmma16(aX, bX2, acc[2]);
    acc[3] = wmma16(aX, bX3, acc[3]);
    aX  = ldfragA(pa + t0);
    bX0 = ldfragB(pb + (size_t)(0 * 16) * TT + t0);
    bX1 = ldfragB(pb + (size_t)(1 * 16) * TT + t0);
    bX2 = ldfragB(pb + (size_t)(2 * 16) * TT + t0);
    bX3 = ldfragB(pb + (size_t)(3 * 16) * TT + t0);
    acc[0] = wmma16(aY, bY0, acc[0]);
    acc[1] = wmma16(aY, bY1, acc[1]);
    acc[2] = wmma16(aY, bY2, acc[2]);
    acc[3] = wmma16(aY, bY3, acc[3]);
    aY  = ldfragA(pa + t0 + 32);
    bY0 = ldfragB(pb + (size_t)(0 * 16) * TT + t0 + 32);
    bY1 = ldfragB(pb + (size_t)(1 * 16) * TT + t0 + 32);
    bY2 = ldfragB(pb + (size_t)(2 * 16) * TT + t0 + 32);
    bY3 = ldfragB(pb + (size_t)(3 * 16) * TT + t0 + 32);
  }
  acc[0] = wmma16(aX, bX0, acc[0]);
  acc[1] = wmma16(aX, bX1, acc[1]);
  acc[2] = wmma16(aX, bX2, acc[2]);
  acc[3] = wmma16(aX, bX3, acc[3]);
  acc[0] = wmma16(aY, bY0, acc[0]);
  acc[1] = wmma16(aY, bY1, acc[1]);
  acc[2] = wmma16(aY, bY2, acc[2]);
  acc[3] = wmma16(aY, bY3, acc[3]);

  const int rTop = (lane >> 4) * 8;
#pragma unroll
  for (int rr = 0; rr < 8; ++rr) {
    const int d = d0 + rr + rTop;
#pragma unroll
    for (int j = 0; j < 4; ++j) {
      const int e = j * 16 + r;
      ctxT[(size_t)bh * (D * D) + e * D + d] = (_Float16)acc[j][rr];  // transposed store
    }
  }
}

// ---------- attn = (q @ ctx) * Dinv + q  (K = 64, WMMA, fused epilogue, f16 out)
// K=64 -> preload all 10 fragments (both k-steps) then 8 WMMAs in one burst.
__global__ void k_qctx(const _Float16* __restrict__ qh, const _Float16* __restrict__ ctxT,
                       const float* __restrict__ Dinv, _Float16* __restrict__ attn) {
  const int lane = threadIdx.x & 31;
  const int h    = threadIdx.x >> 5;      // 8 waves = 8 heads
  const int row0 = blockIdx.x * 16;
  const int b    = row0 / TT;
  const int bh   = b * H + h;
  const int r    = lane & 15;
  const int kA   = (lane >> 4) * 16;
  const int kB   = (lane >> 4) * 16;
  const _Float16* pa = qh + (size_t)(row0 + r) * DIM + h * D + kA;
  const _Float16* pb = ctxT + (size_t)bh * (D * D) + (size_t)r * D + kB;

  const v16h a0  = ldfragA(pa);
  const v16h a1  = ldfragA(pa + 32);
  const v16h b00 = ldfragB(pb + 0 * 16 * D);
  const v16h b01 = ldfragB(pb + 1 * 16 * D);
  const v16h b02 = ldfragB(pb + 2 * 16 * D);
  const v16h b03 = ldfragB(pb + 3 * 16 * D);
  const v16h b10 = ldfragB(pb + 0 * 16 * D + 32);
  const v16h b11 = ldfragB(pb + 1 * 16 * D + 32);
  const v16h b12 = ldfragB(pb + 2 * 16 * D + 32);
  const v16h b13 = ldfragB(pb + 3 * 16 * D + 32);

  v8f acc[4] = {};
  acc[0] = wmma16(a0, b00, acc[0]);
  acc[1] = wmma16(a0, b01, acc[1]);
  acc[2] = wmma16(a0, b02, acc[2]);
  acc[3] = wmma16(a0, b03, acc[3]);
  acc[0] = wmma16(a1, b10, acc[0]);
  acc[1] = wmma16(a1, b11, acc[1]);
  acc[2] = wmma16(a1, b12, acc[2]);
  acc[3] = wmma16(a1, b13, acc[3]);

  const int rTop = (lane >> 4) * 8;
#pragma unroll
  for (int rr = 0; rr < 8; ++rr) {
    const int row = row0 + rr + rTop;
    const float di = Dinv[row * H + h];
#pragma unroll
    for (int j = 0; j < 4; ++j) {
      const int col = h * D + j * 16 + r;
      const float qv = (float)qh[(size_t)row * DIM + col];
      attn[(size_t)row * DIM + col] = (_Float16)(acc[j][rr] * di + qv);
    }
  }
}

// ---------------------------------------------------------------------------
extern "C" void kernel_launch(void* const* d_in, const int* in_sizes, int n_in,
                              void* d_out, int out_size, void* d_ws, size_t ws_size,
                              hipStream_t stream) {
  (void)in_sizes; (void)n_in; (void)out_size; (void)ws_size;
  const float* x  = (const float*)d_in[0];
  const float* Wq = (const float*)d_in[1];
  const float* bq = (const float*)d_in[2];
  const float* Wk = (const float*)d_in[3];
  const float* bk = (const float*)d_in[4];
  const float* Wv = (const float*)d_in[5];
  const float* bv = (const float*)d_in[6];
  const float* Wp = (const float*)d_in[7];
  const float* bp = (const float*)d_in[8];
  float* out = (float*)d_out;

  char* ws = (char*)d_ws;
  const size_t MB = 1ull << 20;
  // Workspace layout with reuse (peak 130 MB):
  _Float16* xh   = (_Float16*)(ws + 0);                    // 16 MB, until GEMM v
  _Float16* wqh  = (_Float16*)(ws + 16 * MB);              // 0.5 MB each
  _Float16* wkh  = (_Float16*)(ws + 16 * MB + 512 * 1024);
  _Float16* wvh  = (_Float16*)(ws + 17 * MB);
  _Float16* wph  = (_Float16*)(ws + 17 * MB + 512 * 1024);
  float*    Pq   = (float*)   (ws + 18 * MB);              // 32 MB raw q proj
  float*    Pk   = (float*)   (ws + 50 * MB);              // 32 MB raw k proj
  _Float16* qh   = (_Float16*)(ws + 82 * MB);              // 16 MB
  _Float16* kh   = (_Float16*)(ws + 98 * MB);              // 16 MB
  _Float16* vh   = (_Float16*)(ws + 114 * MB);             // 16 MB
  _Float16* khT  = (_Float16*)(ws + 18 * MB);              // reuse Pq (free post-softmax)
  _Float16* vhT  = (_Float16*)(ws + 34 * MB);              // reuse Pq upper half
  float*    ksum = (float*)   (ws + 50 * MB);              // reuse Pk (free post-softmax)
  float*    Dinv = (float*)   (ws + 50 * MB + 64 * 1024);
  _Float16* ctxT = (_Float16*)(ws + 51 * MB);
  _Float16* attnh = kh;                                    // reuse kh (free post-transpose)

  // 1) casts to f16
  k_cast_f16<<<(NROW * DIM / 4 + 255) / 256, 256, 0, stream>>>(x, xh, NROW * DIM / 4);
  k_cast_f16<<<(DIM * DIM / 4 + 255) / 256, 256, 0, stream>>>(Wq, wqh, DIM * DIM / 4);
  k_cast_f16<<<(DIM * DIM / 4 + 255) / 256, 256, 0, stream>>>(Wk, wkh, DIM * DIM / 4);
  k_cast_f16<<<(DIM * DIM / 4 + 255) / 256, 256, 0, stream>>>(Wv, wvh, DIM * DIM / 4);
  k_cast_f16<<<(DIM * DIM / 4 + 255) / 256, 256, 0, stream>>>(Wp, wph, DIM * DIM / 4);

  // 2) q/k/v projections (WMMA f16): v straight to f16, q/k raw f32 for softmax
  dim3 gg(NROW / 128, DIM / 64);
  k_gemm_f16<<<gg, 256, 0, stream>>>(xh, wqh, bq, Pq, nullptr);
  k_gemm_f16<<<gg, 256, 0, stream>>>(xh, wkh, bk, Pk, nullptr);
  k_gemm_f16<<<gg, 256, 0, stream>>>(xh, wvh, bv, nullptr, vh);

  // 3) per-head softmax (wave32 shuffle reductions)
  k_softmax_head<<<NROW * H / 8, 256, 0, stream>>>(Pq, qh);
  k_softmax_head<<<NROW * H / 8, 256, 0, stream>>>(Pk, kh);

  // 4) per-head transposes for the K=t GEMMs
  dim3 gt(NROW / 64, H);
  k_transpose_head<<<gt, 256, 0, stream>>>(kh, khT);
  k_transpose_head<<<gt, 256, 0, stream>>>(vh, vhT);

  // 5) k_sum and D_inv
  k_ksum<<<(NB * H * D) / 8, 256, 0, stream>>>(khT, ksum);
  k_dinv<<<NROW * H / 8, 256, 0, stream>>>(qh, ksum, Dinv);

  // 6) context = k^T v (WMMA, K=4096), stored transposed in f16
  k_context<<<NB * H, 128, 0, stream>>>(khT, vhT, ctxT);

  // 7) attn = (q @ ctx) * Dinv + q (WMMA, fused epilogue)
  k_qctx<<<NROW / 16, 256, 0, stream>>>(qh, ctxT, Dinv, attnh);

  // 8) final projection -> d_out (f32)
  k_gemm_f16<<<gg, 256, 0, stream>>>(attnh, wph, bp, out, nullptr);
}
